// VGG_27221502722285
// MI455X (gfx1250) — matrix operand order repaired and checked by reading, not compiled, not run on hardware
//
#include <hip/hip_runtime.h>

typedef _Float16 v8h  __attribute__((ext_vector_type(8)));
typedef _Float16 v16h __attribute__((ext_vector_type(16)));
typedef float    v8f  __attribute__((ext_vector_type(8)));

#define NBATCH 256

// ---------------------------------------------------------------------------
// Input conversion: x fp32 [N=256][3][32][32] -> f16 [32][256][1024],
// channels 3..31 zero-padded so layer-1 needs no channel guard.
// ---------------------------------------------------------------------------
__global__ __launch_bounds__(256) void convert_input(const float* __restrict__ x,
                                                     _Float16* __restrict__ out) {
  int i = blockIdx.x * 256 + threadIdx.x;          // over 32*256*1024
  int c   = i >> 18;                               // / (256*1024)
  int rem = i & ((1 << 18) - 1);
  int n   = rem >> 10;
  int pix = rem & 1023;
  out[i] = (c < 3) ? (_Float16)x[(n * 3 + c) * 1024 + pix] : (_Float16)0.f;
}

// ---------------------------------------------------------------------------
// Fold conv bias + eval-mode BN into per-channel scale/shift.
// ---------------------------------------------------------------------------
__global__ void prep_bn(const float* __restrict__ b, const float* __restrict__ g,
                        const float* __restrict__ be, const float* __restrict__ mean,
                        const float* __restrict__ var,
                        float* __restrict__ scale, float* __restrict__ shift, int C) {
  int c = blockIdx.x * blockDim.x + threadIdx.x;
  if (c >= C) return;
  float inv = rsqrtf(var[c] + 1e-5f);
  float s = g[c] * inv;
  scale[c] = s;
  shift[c] = be[c] - mean[c] * s + b[c] * s;
}

// ---------------------------------------------------------------------------
// Pack weights fp32 [Cout][Cin][3][3] -> f16 [Cout][9][Cinp] (tap-major),
// channel dim zero-padded Cin -> Cinp (multiple of 32).
// ---------------------------------------------------------------------------
__global__ void pack_w(const float* __restrict__ w, _Float16* __restrict__ wp,
                       int Cin, int Cinp, int total) {
  int i = blockIdx.x * blockDim.x + threadIdx.x;   // over Cout*9*Cinp
  if (i >= total) return;
  int Kp2 = 9 * Cinp;
  int co = i / Kp2;
  int t  = i - co * Kp2;
  int r  = t / Cinp;
  int ci = t - r * Cinp;
  wp[i] = (ci < Cin) ? (_Float16)w[(co * Cin + ci) * 9 + r] : (_Float16)0.f;
}

// ---------------------------------------------------------------------------
// Conv3x3 SAME as 9 shifted GEMMs + scale/shift + ReLU.  f16 in/out, f32 acc.
// Block = 256 threads (8 waves).  Tile: M = WM*16, N = (8/WM)*SUB*16.
// Double-buffered LDS B tile [pixel][k] (row stride 40), one barrier/K-step:
// next step's global loads (B fill + A slab) overlap current step's WMMAs.
// ---------------------------------------------------------------------------
template <int WM, int SUB>
__global__ __launch_bounds__(256) void conv_gemm_wmma(
    const _Float16* __restrict__ actIn, const _Float16* __restrict__ Wp,
    const float* __restrict__ scale, const float* __restrict__ shift,
    _Float16* __restrict__ actOut,
    int Cinp, int H, int NHW) {
  constexpr int WN   = 8 / WM;
  constexpr int NT   = WN * SUB * 16;            // tile width (pixels)
  constexpr int ROWS = 40;                       // padded LDS row (f16 units)
  constexpr int CPT  = NT / 64;                  // v8h chunks per thread
  constexpr int CSTR = 256 / NT;                 // chunk stride between threads
  __shared__ _Float16 bt[2][NT * ROWS];

  const int tid  = threadIdx.x;
  const int lane = tid & 31;
  const int wave = tid >> 5;
  const int mwave = wave % WM;
  const int nwave = wave / WM;
  const int m0 = blockIdx.y * (WM * 16) + mwave * 16;
  const int colbase = nwave * SUB * 16;
  const int p0 = blockIdx.x * NT;
  const int HW = H * H;
  const int Kp2 = 9 * Cinp;

  // fill-side coordinates: this thread owns im2col pixel column pp0
  const int pp0   = tid % NT;
  const int cbase = tid / NT;
  const int p = p0 + pp0;
  const int n = p / HW;
  int rem = p - n * HW;
  const int y = rem / H;
  const int x = rem - y * H;

  v8f acc[SUB] = {};

  // A-fragment per-lane addressing (ISA 16-bit A 16x32 layout)
  const _Float16* wrow = Wp + (size_t)(m0 + (lane & 15)) * Kp2 + (lane >> 4) * 8;
  const int kb = (lane >> 4) * 8;                // B-fragment K offset

  // branch-free tap state: shifted base address + spatial predicate
  auto tapState = [&](int r, int& gbase, bool& pred) {
    int dy = r / 3, dx = r - (r / 3) * 3;
    int off = (dy - 1) * H + (dx - 1);
    pred = ((unsigned)(y + dy - 1) < (unsigned)H) &&
           ((unsigned)(x + dx - 1) < (unsigned)H);
    gbase = pred ? (p + off) : p;                // always in-range address
  };

  auto fill = [&](int buf, int gbase, bool pred, int k0) {
    v8h zv = {};
#pragma unroll
    for (int i2 = 0; i2 < CPT; ++i2) {
      int chunk = cbase + i2 * CSTR;
      int kkb = chunk * 8;
      v8h vv;
#pragma unroll
      for (int j = 0; j < 8; ++j)
        vv[j] = actIn[(size_t)(k0 + kkb + j) * NHW + gbase];
      vv = pred ? vv : zv;
      *reinterpret_cast<v8h*>(&bt[buf][pp0 * ROWS + kkb]) = vv;
    }
  };

  const int KCH = Cinp >> 5;                     // K-steps per tap
  const int KSTEPS = 9 * KCH;

  int r = 0, kc = 0, gbase;
  bool pred;
  tapState(0, gbase, pred);
  fill(0, gbase, pred, 0);
  v8h alo = *reinterpret_cast<const v8h*>(wrow);
  v8h ahi = *reinterpret_cast<const v8h*>(wrow + 16);
  __syncthreads();

  int cur = 0;
  for (int s = 0; s < KSTEPS; ++s) {
    int rn = r, kcn = kc + 1;
    if (kcn == KCH) { kcn = 0; rn = r + 1; }
    v8h nlo = alo, nhi = ahi;
    if (s + 1 < KSTEPS) {
      int gb; bool pr;
      tapState(rn, gb, pr);
      fill(cur ^ 1, gb, pr, kcn * 32);           // next B tile in flight
      const _Float16* wn = wrow + rn * Cinp + kcn * 32;
      nlo = *reinterpret_cast<const v8h*>(wn);   // next A slab in flight
      nhi = *reinterpret_cast<const v8h*>(wn + 16);
    }

    v16h a = __builtin_shufflevector(alo, ahi, 0, 1, 2, 3, 4, 5, 6, 7,
                                     8, 9, 10, 11, 12, 13, 14, 15);
#pragma unroll
    for (int c0 = 0; c0 < SUB; ++c0) {
      const _Float16* bp = &bt[cur][(colbase + c0 * 16 + (lane & 15)) * ROWS + kb];
      v8h blo = *reinterpret_cast<const v8h*>(bp);
      v8h bhi = *reinterpret_cast<const v8h*>(bp + 16);
      v16h b = __builtin_shufflevector(blo, bhi, 0, 1, 2, 3, 4, 5, 6, 7,
                                       8, 9, 10, 11, 12, 13, 14, 15);
      acc[c0] = __builtin_amdgcn_wmma_f32_16x16x32_f16(
          false, a, false, b, (short)0, acc[c0], false, false);
    }
    __syncthreads();
    alo = nlo; ahi = nhi;
    r = rn; kc = kcn; cur ^= 1;
  }

  // ---- epilogue: scale/shift + ReLU, store f16 (C/D layout: M=rr+8*half) ----
  const int nloc = lane & 15;
  const int mh = (lane >> 4) * 8;
#pragma unroll
  for (int c0 = 0; c0 < SUB; ++c0) {
    int pp = p0 + colbase + c0 * 16 + nloc;
#pragma unroll
    for (int rr = 0; rr < 8; ++rr) {
      int ch = m0 + mh + rr;
      float v = acc[c0][rr] * scale[ch] + shift[ch];
      actOut[(size_t)ch * NHW + pp] = (_Float16)(v > 0.f ? v : 0.f);
    }
  }
}

// ---------------------------------------------------------------------------
// 2x2 maxpool, stride 2, f16 [C][256][H][H] -> [C][256][H/2][H/2]
// ---------------------------------------------------------------------------
__global__ __launch_bounds__(256) void maxpool2(const _Float16* __restrict__ in,
                                                _Float16* __restrict__ out,
                                                int C, int H) {
  int Ho = H >> 1;
  int total = C * NBATCH * Ho * Ho;
  for (int i = blockIdx.x * blockDim.x + threadIdx.x; i < total;
       i += gridDim.x * blockDim.x) {
    int xo = i % Ho;
    int t  = i / Ho;
    int yo = t % Ho;
    t /= Ho;                                     // t = c*256 + n
    const _Float16* ptr = in + ((size_t)t * H + yo * 2) * H + xo * 2;
    float a = (float)ptr[0], b = (float)ptr[1];
    float c = (float)ptr[H], d = (float)ptr[H + 1];
    out[i] = (_Float16)fmaxf(fmaxf(a, b), fmaxf(c, d));
  }
}

// ---------------------------------------------------------------------------
// FC: act f16 [512][256] -> out fp32 [256][10]
// ---------------------------------------------------------------------------
__global__ __launch_bounds__(256) void fc10(const _Float16* __restrict__ act,
                                            const float* __restrict__ w,
                                            const float* __restrict__ b,
                                            float* __restrict__ out) {
  int n = threadIdx.x;                           // 0..255
  int j = blockIdx.x;                            // 0..9
  float s = b[j];
  for (int c = 0; c < 512; ++c)
    s += (float)act[c * NBATCH + n] * w[j * 512 + c];
  out[n * 10 + j] = s;
}

// ---------------------------------------------------------------------------
extern "C" void kernel_launch(void* const* d_in, const int* in_sizes, int n_in,
                              void* d_out, int out_size, void* d_ws, size_t ws_size,
                              hipStream_t stream) {
  (void)n_in; (void)out_size; (void)ws_size;

  struct CP { const float *w, *b, *g, *be, *m, *v; };
  CP cp[13];
  const float *x, *fcw, *fcb;

  // Detect flattening order: insertion order has x (256*3*32*32) first.
  if (in_sizes[0] == 3 * 256 * 32 * 32) {
    x = (const float*)d_in[0];
    int idx = 1;
    for (int l = 0; l < 13; ++l, idx += 6) {
      cp[l].w  = (const float*)d_in[idx + 0];
      cp[l].b  = (const float*)d_in[idx + 1];
      cp[l].g  = (const float*)d_in[idx + 2];
      cp[l].be = (const float*)d_in[idx + 3];
      cp[l].m  = (const float*)d_in[idx + 4];
      cp[l].v  = (const float*)d_in[idx + 5];
    }
    fcw = (const float*)d_in[79];
    fcb = (const float*)d_in[80];
  } else {
    // pytree-sorted: convs(b,beta,gamma,mean,var,w)*13, fc_b, fc_w, x
    int idx = 0;
    for (int l = 0; l < 13; ++l, idx += 6) {
      cp[l].b  = (const float*)d_in[idx + 0];
      cp[l].be = (const float*)d_in[idx + 1];
      cp[l].g  = (const float*)d_in[idx + 2];
      cp[l].m  = (const float*)d_in[idx + 3];
      cp[l].v  = (const float*)d_in[idx + 4];
      cp[l].w  = (const float*)d_in[idx + 5];
    }
    fcb = (const float*)d_in[78];
    fcw = (const float*)d_in[79];
    x   = (const float*)d_in[80];
  }

  // ---- workspace carve-up (256B aligned) ----
  char* ws = (char*)d_ws;
  size_t off = 0;
  auto alloc = [&](size_t bytes) {
    void* p = ws + off;
    off += (bytes + 255) & ~(size_t)255;
    return p;
  };
  _Float16* bufA = (_Float16*)alloc((size_t)64 * 256 * 1024 * 2);  // 33.5 MB
  _Float16* bufB = (_Float16*)alloc((size_t)64 * 256 * 1024 * 2);  // 33.5 MB
  _Float16* wp   = (_Float16*)alloc((size_t)512 * 9 * 512 * 2);    // 4.7 MB
  float* scale   = (float*)alloc(512 * 4);
  float* shift   = (float*)alloc(512 * 4);

  convert_input<<<(32 * 256 * 1024) / 256, 256, 0, stream>>>(x, bufA);

  static const int  cins[13]  = {3, 64, 64, 128, 128, 256, 256, 256, 512, 512, 512, 512, 512};
  static const int  couts[13] = {64, 64, 128, 128, 256, 256, 256, 512, 512, 512, 512, 512, 512};
  static const bool pool[13]  = {false, true, false, true, false, false, true,
                                 false, false, true, false, false, true};

  _Float16* cur = bufA;
  _Float16* nxt = bufB;
  int H = 32;
  for (int l = 0; l < 13; ++l) {
    int Cin = cins[l], Cout = couts[l];
    int Cinp = (Cin + 31) & ~31;
    int NHW = 256 * H * H;

    prep_bn<<<(Cout + 63) / 64, 64, 0, stream>>>(cp[l].b, cp[l].g, cp[l].be,
                                                 cp[l].m, cp[l].v, scale, shift, Cout);
    int tot = Cout * 9 * Cinp;
    pack_w<<<(tot + 255) / 256, 256, 0, stream>>>(cp[l].w, wp, Cin, Cinp, tot);

    if (Cout < 128) {                            // 64x256 tile
      dim3 grid(NHW / 256, Cout / 64);
      conv_gemm_wmma<4, 8><<<grid, 256, 0, stream>>>(cur, wp, scale, shift, nxt,
                                                     Cinp, H, NHW);
    } else if (NHW >= 8192) {                    // 128x128 tile
      dim3 grid(NHW / 128, Cout / 128);
      conv_gemm_wmma<8, 8><<<grid, 256, 0, stream>>>(cur, wp, scale, shift, nxt,
                                                     Cinp, H, NHW);
    } else {                                     // 128x64 tile (tail layers)
      dim3 grid(NHW / 64, Cout / 128);
      conv_gemm_wmma<8, 4><<<grid, 256, 0, stream>>>(cur, wp, scale, shift, nxt,
                                                     Cinp, H, NHW);
    }
    { _Float16* t = cur; cur = nxt; nxt = t; }

    if (pool[l]) {
      int Ho = H / 2;
      int total = Cout * 256 * Ho * Ho;
      int blocks = (total + 255) / 256;
      if (blocks > 4096) blocks = 4096;
      maxpool2<<<blocks, 256, 0, stream>>>(cur, nxt, Cout, H);
      { _Float16* t = cur; cur = nxt; nxt = t; }
      H = Ho;
    }
  }

  fc10<<<10, 256, 0, stream>>>(cur, fcw, fcb, (float*)d_out);
}
